// SelfAttention_15444702396493
// MI455X (gfx1250) — matrix operand order repaired and checked by reading, not compiled, not run on hardware
//
#include <hip/hip_runtime.h>
#include <stdint.h>

typedef __attribute__((ext_vector_type(16))) _Float16 v16h;
typedef __attribute__((ext_vector_type(8)))  _Float16 v8h;
typedef __attribute__((ext_vector_type(4)))  _Float16 v4h;
typedef __attribute__((ext_vector_type(8)))  float    v8f;
typedef __attribute__((ext_vector_type(4)))  float    f32x4;

#define B_DIM 2
#define H_DIM 16
#define S_DIM 2048
#define D_DIM 64
#define BQA 128        // async path: query rows per block (32 per wave, 2 A-tiles)
#define BQF 64         // fallback: query rows per block (16 per wave)
#define BK 64          // keys per iteration
#define NWAVES 4
#define LDK 72         // padded f16 row stride (bank-conflict-free)
#define LDV 72
#define LDP 72
#define KTILE_BYTES (BK * LDK * 2)     // 9216

__device__ __forceinline__ v16h h16cat(v8h lo, v8h hi) {
  return __builtin_shufflevector(lo, hi, 0,1,2,3,4,5,6,7,8,9,10,11,12,13,14,15);
}

// ---- CDNA5 async copy: global -> LDS, 16B per lane, tracked by ASYNCcnt ----
__device__ __forceinline__ void async_b128(uint32_t lds_off, uint32_t voff,
                                           const void* sbase) {
  asm volatile("global_load_async_to_lds_b128 %0, %1, %2"
               :: "v"(lds_off), "v"(voff), "s"(sbase) : "memory");
}

// =====================  pre-pass kernels  =====================

__global__ __launch_bounds__(256) void cvt_f16(const float* __restrict__ src,
                                               _Float16* __restrict__ dst,
                                               float scale) {
  size_t i = ((size_t)blockIdx.x * 256 + threadIdx.x) * 8;
  f32x4 x0 = *(const f32x4*)(src + i);
  f32x4 x1 = *(const f32x4*)(src + i + 4);
  v4h a, b;
  #pragma unroll
  for (int t = 0; t < 4; ++t) {
    a[t] = (_Float16)(x0[t] * scale);
    b[t] = (_Float16)(x1[t] * scale);
  }
  *(v4h*)(dst + i)     = a;
  *(v4h*)(dst + i + 4) = b;
}

// V: [bh][s][f] f32  ->  Vt: [bh][f][s] f16  (64x64 tiles through LDS)
__global__ __launch_bounds__(128) void vtranspose_f16(const float* __restrict__ V,
                                                      _Float16* __restrict__ Vt) {
  __shared__ _Float16 sT[D_DIM * LDV];
  const int tid = threadIdx.x;
  const int bh = blockIdx.x >> 5;
  const int kb = blockIdx.x & 31;
  const int kg = tid >> 3;
  const int fg = tid & 7;

  const float* vp = V + ((size_t)bh * S_DIM + kb * BK) * D_DIM;
  float vr[4][8];
  #pragma unroll
  for (int j = 0; j < 4; ++j) {
    const float* src = vp + (size_t)(4 * kg + j) * D_DIM + 8 * fg;
    f32x4 x0 = *(const f32x4*)(src);
    f32x4 x1 = *(const f32x4*)(src + 4);
    #pragma unroll
    for (int t = 0; t < 4; ++t) { vr[j][t] = x0[t]; vr[j][4 + t] = x1[t]; }
  }
  #pragma unroll
  for (int f = 0; f < 8; ++f) {
    v4h pk;
    #pragma unroll
    for (int j = 0; j < 4; ++j) pk[j] = (_Float16)vr[j][f];
    *(v4h*)&sT[(8 * fg + f) * LDV + 4 * kg] = pk;
  }
  __syncthreads();
  _Float16* dst = Vt + (size_t)bh * D_DIM * S_DIM + kb * BK;
  #pragma unroll
  for (int t = 0; t < 4; ++t) {
    int f = kg + 16 * t;
    v8h x = *(const v8h*)&sT[f * LDV + fg * 8];
    *(v8h*)(dst + (size_t)f * S_DIM + fg * 8) = x;
  }
}

// =====  main attention kernel: async double-buffered, 32 q-rows per wave  =====

__global__ __launch_bounds__(128) void fa_fwd_async(
    const _Float16* __restrict__ Qh, const _Float16* __restrict__ Kh,
    const _Float16* __restrict__ Vth, float* __restrict__ O)
{
  __shared__ _Float16 sK [2][BK * LDK];     // [key][feat] f16
  __shared__ _Float16 sVt[2][D_DIM * LDV];  // [feat][key] f16
  __shared__ _Float16 sP [NWAVES * 32 * LDP];

  const int tid  = threadIdx.x;
  const int wave = tid >> 5;
  const int lane = tid & 31;
  const int ln   = lane & 15;
  const int hi   = lane >> 4;

  const int qblocks = S_DIM / BQA;            // 16
  const int bh = blockIdx.x / qblocks;
  const int qb = blockIdx.x % qblocks;
  const int b  = bh / H_DIM;
  const int h  = bh % H_DIM;

  const _Float16* kbh  = Kh  + (size_t)bh * S_DIM * D_DIM;
  const _Float16* vtbh = Vth + (size_t)bh * D_DIM * S_DIM;

  const int q0 = qb * BQA + wave * 32;        // 32 q-rows per wave

  // ---- Q tiles (already scaled f16) into WMMA A-layout: 2 M-tiles x 2 K-chunks ----
  v16h aq[2][2];
  #pragma unroll
  for (int t = 0; t < 2; ++t) {
    const _Float16* qp = Qh + ((size_t)bh * S_DIM + q0 + t * 16 + ln) * D_DIM;
    #pragma unroll
    for (int kk = 0; kk < 2; ++kk) {
      v8h lo = *(const v8h*)(qp + kk * 32 + hi * 8);
      v8h hb = *(const v8h*)(qp + kk * 32 + hi * 8 + 16);
      aq[t][kk] = h16cat(lo, hb);
    }
  }

  v8f oacc[2][4];
  #pragma unroll
  for (int t = 0; t < 2; ++t)
    #pragma unroll
    for (int s = 0; s < 4; ++s)
      #pragma unroll
      for (int r = 0; r < 8; ++r) oacc[t][s][r] = 0.0f;

  float mrow[2][8], lpart[2][8];
  #pragma unroll
  for (int t = 0; t < 2; ++t)
    #pragma unroll
    for (int r = 0; r < 8; ++r) { mrow[t][r] = -1.0e30f; lpart[t][r] = 0.0f; }

  const int kg = tid >> 3;
  const int fg = tid & 7;

  const uint32_t sK_lds  = (uint32_t)(uintptr_t)&sK[0][0];
  const uint32_t sVt_lds = (uint32_t)(uintptr_t)&sVt[0][0];

  #define ISSUE_TILE(KB, BUF)                                                   \
    {                                                                           \
      _Pragma("unroll")                                                         \
      for (int t = 0; t < 4; ++t) {                                             \
        uint32_t r = (uint32_t)(kg + 16 * t);                                   \
        async_b128(sK_lds  + (BUF) * KTILE_BYTES + r * 144u + fg * 16u,         \
                   (uint32_t)((KB) * 8192 + r * 128u + fg * 16u), kbh);         \
        async_b128(sVt_lds + (BUF) * KTILE_BYTES + r * 144u + fg * 16u,         \
                   (uint32_t)(r * 4096u + (KB) * 128 + fg * 16u), vtbh);        \
      }                                                                         \
    }

  ISSUE_TILE(0, 0)

  for (int kb = 0; kb < S_DIM / BK; ++kb) {
    const int buf = kb & 1;
    asm volatile("s_wait_asynccnt 0x0" ::: "memory");
    __syncthreads();                        // tile kb visible to whole block
    if (kb + 1 < S_DIM / BK) ISSUE_TILE(kb + 1, buf ^ 1)

    const _Float16* sKb  = &sK [buf][0];
    const _Float16* sVtb = &sVt[buf][0];

    // ---- S = Q * K^T : each B fragment feeds BOTH A tiles ----
    v8f sacc[2][4];
    #pragma unroll
    for (int sub = 0; sub < 4; ++sub) {
      v8f a0 = {0.f,0.f,0.f,0.f,0.f,0.f,0.f,0.f};
      v8f a1 = {0.f,0.f,0.f,0.f,0.f,0.f,0.f,0.f};
      #pragma unroll
      for (int kk = 0; kk < 2; ++kk) {
        const _Float16* bp = &sKb[(sub * 16 + ln) * LDK + kk * 32 + hi * 8];
        v16h bk = h16cat(*(const v8h*)bp, *(const v8h*)(bp + 16));
        a0 = __builtin_amdgcn_wmma_f32_16x16x32_f16(false, aq[0][kk], false, bk,
                                                    (short)0, a0, false, false);
        a1 = __builtin_amdgcn_wmma_f32_16x16x32_f16(false, aq[1][kk], false, bk,
                                                    (short)0, a1, false, false);
      }
      sacc[0][sub] = a0;
      sacc[1][sub] = a1;
    }

    // ---- online softmax (base-2), per M-tile; l kept as per-lane partials ----
    #pragma unroll
    for (int t = 0; t < 2; ++t) {
      float mnew[8];
      #pragma unroll
      for (int r = 0; r < 8; ++r) {
        float mx = mrow[t][r];
        #pragma unroll
        for (int sub = 0; sub < 4; ++sub) mx = fmaxf(mx, sacc[t][sub][r]);
        #pragma unroll
        for (int d = 1; d <= 8; d <<= 1) mx = fmaxf(mx, __shfl_xor(mx, d, 32));
        mnew[r] = mx;
      }
      #pragma unroll
      for (int r = 0; r < 8; ++r) {
        float alpha = __builtin_amdgcn_exp2f(mrow[t][r] - mnew[r]);
        mrow[t][r] = mnew[r];
        float ps = 0.0f;
        #pragma unroll
        for (int sub = 0; sub < 4; ++sub) {
          float p = __builtin_amdgcn_exp2f(sacc[t][sub][r] - mnew[r]);
          sacc[t][sub][r] = p;
          ps += p;
        }
        lpart[t][r] = lpart[t][r] * alpha + ps;
        #pragma unroll
        for (int sub = 0; sub < 4; ++sub) oacc[t][sub][r] *= alpha;
      }
    }

    // ---- stage P: C/D layout -> LDS -> A layout (wave-private, in-order DS) ----
    _Float16* pw = &sP[wave * 32 * LDP];
    #pragma unroll
    for (int t = 0; t < 2; ++t)
      #pragma unroll
      for (int sub = 0; sub < 4; ++sub)
        #pragma unroll
        for (int r = 0; r < 8; ++r)
          pw[(t * 16 + r + 8 * hi) * LDP + sub * 16 + ln] =
              (_Float16)sacc[t][sub][r];

    asm volatile("s_wait_dscnt 0" ::: "memory");

    v16h ap[2][2];
    #pragma unroll
    for (int t = 0; t < 2; ++t)
      #pragma unroll
      for (int c = 0; c < 2; ++c) {
        const _Float16* pp = &pw[(t * 16 + ln) * LDP + c * 32 + hi * 8];
        ap[t][c] = h16cat(*(const v8h*)pp, *(const v8h*)(pp + 16));
      }

    // ---- O += P * V : each B fragment feeds BOTH A tiles ----
    #pragma unroll
    for (int sub = 0; sub < 4; ++sub) {
      v8f a0 = oacc[0][sub];
      v8f a1 = oacc[1][sub];
      #pragma unroll
      for (int c = 0; c < 2; ++c) {
        const _Float16* bp = &sVtb[(sub * 16 + ln) * LDV + c * 32 + hi * 8];
        v16h bv = h16cat(*(const v8h*)bp, *(const v8h*)(bp + 16));
        a0 = __builtin_amdgcn_wmma_f32_16x16x32_f16(false, ap[0][c], false, bv,
                                                    (short)0, a0, false, false);
        a1 = __builtin_amdgcn_wmma_f32_16x16x32_f16(false, ap[1][c], false, bv,
                                                    (short)0, a1, false, false);
      }
      oacc[0][sub] = a0;
      oacc[1][sub] = a1;
    }
  }

  // ---- epilogue: one cross-lane l reduction, normalize, permuted store ----
  #pragma unroll
  for (int t = 0; t < 2; ++t)
    #pragma unroll
    for (int r = 0; r < 8; ++r) {
      float l = lpart[t][r];
      #pragma unroll
      for (int d = 1; d <= 8; d <<= 1) l += __shfl_xor(l, d, 32);
      float inv = __builtin_amdgcn_rcpf(l);
      int grow = q0 + t * 16 + r + 8 * hi;
      #pragma unroll
      for (int sub = 0; sub < 4; ++sub) {
        O[((size_t)(b * S_DIM + grow) * H_DIM + h) * D_DIM + sub * 16 + ln] =
            oacc[t][sub][r] * inv;
      }
    }
  #undef ISSUE_TILE
}

// =====================  fallback (self-contained, no workspace)  =====================

__global__ __launch_bounds__(128) void fa_fwd_f16wmma(
    const float* __restrict__ Q, const float* __restrict__ K,
    const float* __restrict__ V, float* __restrict__ O)
{
  __shared__ _Float16 sK [BK    * LDK];
  __shared__ _Float16 sVt[D_DIM * LDV];
  __shared__ _Float16 sP [NWAVES * 16 * LDP];

  const int tid  = threadIdx.x;
  const int wave = tid >> 5;
  const int lane = tid & 31;
  const int ln   = lane & 15;
  const int hi   = lane >> 4;

  const int qblocks = S_DIM / BQF;
  const int bh = blockIdx.x / qblocks;
  const int qb = blockIdx.x % qblocks;
  const int b  = bh / H_DIM;
  const int h  = bh % H_DIM;

  const float* qbase = Q + (size_t)bh * S_DIM * D_DIM;
  const float* kbase = K + (size_t)bh * S_DIM * D_DIM;
  const float* vbase = V + (size_t)bh * S_DIM * D_DIM;

  const int q0   = qb * BQF + wave * 16;
  const int qrow = q0 + ln;

  const float qscale = 0.125f * 1.44269504088896340736f;
  v16h aq[2];
  #pragma unroll
  for (int kk = 0; kk < 2; ++kk) {
    const float* qr = qbase + (size_t)qrow * D_DIM + kk * 32 + hi * 8;
    #pragma unroll
    for (int g = 0; g < 2; ++g) {
      f32x4 x0 = *(const f32x4*)(qr + g * 16);
      f32x4 x1 = *(const f32x4*)(qr + g * 16 + 4);
      #pragma unroll
      for (int j = 0; j < 4; ++j) {
        aq[kk][g * 8 + j]     = (_Float16)(x0[j] * qscale);
        aq[kk][g * 8 + 4 + j] = (_Float16)(x1[j] * qscale);
      }
    }
  }

  v8f oacc[4];
  #pragma unroll
  for (int s = 0; s < 4; ++s)
    #pragma unroll
    for (int r = 0; r < 8; ++r) oacc[s][r] = 0.0f;

  float mrow[8], lpart[8];
  #pragma unroll
  for (int r = 0; r < 8; ++r) { mrow[r] = -1.0e30f; lpart[r] = 0.0f; }

  const int kg = tid >> 3;
  const int fg = tid & 7;

  for (int kb = 0; kb < S_DIM / BK; ++kb) {
    __syncthreads();
    {
      const float* kp = kbase + (size_t)(kb * BK) * D_DIM;
      const float* vp = vbase + (size_t)(kb * BK) * D_DIM;
      #pragma unroll
      for (int j = 0; j < 4; ++j) {
        const float* src = kp + (size_t)(4 * kg + j) * D_DIM + 8 * fg;
        f32x4 x0 = *(const f32x4*)(src);
        f32x4 x1 = *(const f32x4*)(src + 4);
        v4h h0, h1;
        #pragma unroll
        for (int t = 0; t < 4; ++t) { h0[t] = (_Float16)x0[t]; h1[t] = (_Float16)x1[t]; }
        *(v4h*)&sK[(4 * kg + j) * LDK + 8 * fg]     = h0;
        *(v4h*)&sK[(4 * kg + j) * LDK + 8 * fg + 4] = h1;
      }
      float vr[4][8];
      #pragma unroll
      for (int j = 0; j < 4; ++j) {
        const float* src = vp + (size_t)(4 * kg + j) * D_DIM + 8 * fg;
        f32x4 x0 = *(const f32x4*)(src);
        f32x4 x1 = *(const f32x4*)(src + 4);
        #pragma unroll
        for (int t = 0; t < 4; ++t) { vr[j][t] = x0[t]; vr[j][4 + t] = x1[t]; }
      }
      #pragma unroll
      for (int f = 0; f < 8; ++f) {
        v4h pk;
        #pragma unroll
        for (int j = 0; j < 4; ++j) pk[j] = (_Float16)vr[j][f];
        *(v4h*)&sVt[(8 * fg + f) * LDV + 4 * kg] = pk;
      }
    }
    __syncthreads();

    v8f sacc[4];
    #pragma unroll
    for (int sub = 0; sub < 4; ++sub) {
      v8f acc = {0.f,0.f,0.f,0.f,0.f,0.f,0.f,0.f};
      #pragma unroll
      for (int kk = 0; kk < 2; ++kk) {
        const _Float16* bp = &sK[(sub * 16 + ln) * LDK + kk * 32 + hi * 8];
        v16h bk = h16cat(*(const v8h*)bp, *(const v8h*)(bp + 16));
        acc = __builtin_amdgcn_wmma_f32_16x16x32_f16(false, aq[kk], false, bk,
                                                     (short)0, acc, false, false);
      }
      sacc[sub] = acc;
    }

    float mnew[8];
    #pragma unroll
    for (int r = 0; r < 8; ++r) {
      float mx = mrow[r];
      #pragma unroll
      for (int sub = 0; sub < 4; ++sub) mx = fmaxf(mx, sacc[sub][r]);
      #pragma unroll
      for (int d = 1; d <= 8; d <<= 1) mx = fmaxf(mx, __shfl_xor(mx, d, 32));
      mnew[r] = mx;
    }
    #pragma unroll
    for (int r = 0; r < 8; ++r) {
      float alpha = __builtin_amdgcn_exp2f(mrow[r] - mnew[r]);
      mrow[r] = mnew[r];
      float ps = 0.0f;
      #pragma unroll
      for (int sub = 0; sub < 4; ++sub) {
        float p = __builtin_amdgcn_exp2f(sacc[sub][r] - mnew[r]);
        sacc[sub][r] = p;
        ps += p;
      }
      lpart[r] = lpart[r] * alpha + ps;
      #pragma unroll
      for (int sub = 0; sub < 4; ++sub) oacc[sub][r] *= alpha;
    }

    _Float16* pw = &sP[wave * 16 * LDP];
    #pragma unroll
    for (int sub = 0; sub < 4; ++sub)
      #pragma unroll
      for (int r = 0; r < 8; ++r)
        pw[(r + 8 * hi) * LDP + sub * 16 + ln] = (_Float16)sacc[sub][r];

    asm volatile("s_wait_dscnt 0" ::: "memory");

    v16h ap[2];
    #pragma unroll
    for (int c = 0; c < 2; ++c) {
      const _Float16* pp = &pw[ln * LDP + c * 32 + hi * 8];
      ap[c] = h16cat(*(const v8h*)pp, *(const v8h*)(pp + 16));
    }

    #pragma unroll
    for (int sub = 0; sub < 4; ++sub) {
      v8f acc = oacc[sub];
      #pragma unroll
      for (int c = 0; c < 2; ++c) {
        const _Float16* bp = &sVt[(sub * 16 + ln) * LDV + c * 32 + hi * 8];
        v16h bv = h16cat(*(const v8h*)bp, *(const v8h*)(bp + 16));
        acc = __builtin_amdgcn_wmma_f32_16x16x32_f16(false, ap[c], false, bv,
                                                     (short)0, acc, false, false);
      }
      oacc[sub] = acc;
    }
  }

  #pragma unroll
  for (int r = 0; r < 8; ++r) {
    float l = lpart[r];
    #pragma unroll
    for (int d = 1; d <= 8; d <<= 1) l += __shfl_xor(l, d, 32);
    float inv = __builtin_amdgcn_rcpf(l);
    int grow = q0 + r + 8 * hi;
    #pragma unroll
    for (int sub = 0; sub < 4; ++sub) {
      O[((size_t)(b * S_DIM + grow) * H_DIM + h) * D_DIM + sub * 16 + ln] =
          oacc[sub][r] * inv;
    }
  }
}

extern "C" void kernel_launch(void* const* d_in, const int* in_sizes, int n_in,
                              void* d_out, int out_size, void* d_ws, size_t ws_size,
                              hipStream_t stream) {
  const float* q = (const float*)d_in[0];
  const float* k = (const float*)d_in[1];
  const float* v = (const float*)d_in[2];
  float* out = (float*)d_out;
  (void)in_sizes; (void)n_in; (void)out_size;

  const size_t n = (size_t)B_DIM * H_DIM * S_DIM * D_DIM;  // 4,194,304
  const size_t need = 3 * n * sizeof(_Float16);            // 24 MB
  const float qscale = 0.125f * 1.44269504088896340736f;

  if (ws_size >= need) {
    _Float16* qh = (_Float16*)d_ws;
    _Float16* kh = qh + n;
    _Float16* vt = kh + n;
    const int cvt_blocks = (int)(n / (256 * 8));            // 2048
    cvt_f16<<<cvt_blocks, 256, 0, stream>>>(q, qh, qscale);
    cvt_f16<<<cvt_blocks, 256, 0, stream>>>(k, kh, 1.0f);
    vtranspose_f16<<<B_DIM * H_DIM * (S_DIM / BK), 128, 0, stream>>>(v, vt);
    fa_fwd_async<<<dim3(B_DIM * H_DIM * (S_DIM / BQA)), dim3(128), 0, stream>>>(
        qh, kh, vt, out);
  } else {
    fa_fwd_f16wmma<<<dim3(B_DIM * H_DIM * (S_DIM / BQF)), dim3(128), 0, stream>>>(
        q, k, v, out);
  }
}
